// SelfAttention_v2_41102837023261
// MI455X (gfx1250) — matrix operand order repaired and checked by reading, not compiled, not run on hardware
//
#include <hip/hip_runtime.h>
#include <hip/hip_bf16.h>

#define BDIM 256

typedef __bf16 bf16_t;
typedef __attribute__((ext_vector_type(16))) __bf16 v16bf;
typedef __attribute__((ext_vector_type(8)))  __bf16 v8bf;
typedef __attribute__((ext_vector_type(8)))  float  v8f;
typedef __attribute__((ext_vector_type(4)))  unsigned v4u;
typedef __attribute__((ext_vector_type(8)))  unsigned v8u;

static constexpr int Bn   = 2;
static constexpr int Sn   = 4096;
static constexpr int Dn   = 1024;
static constexpr int Mtot = Bn * Sn;     // 8192 total rows

// ---------------- float -> bf16: native RNE convert (v_cvt_*_bf16_f32) ----------------
__device__ __forceinline__ bf16_t f2bf(float f) { return (bf16_t)f; }

// ---------------- WMMA fragment loaders (contiguous k = 16*(lane/16) + s, same order for B) ----------------
// A 16x32 bf16: lane holds row M=lane%16; K chunk [k0..k0+15], k0=(lane/16)*16
__device__ __forceinline__ v16bf load_a_frag(const bf16_t* __restrict__ base, int ld) {
  const int lane = threadIdx.x & 31;
  const bf16_t* p = base + (size_t)(lane & 15) * ld + ((lane >> 4) << 4);
  v8bf lo = *reinterpret_cast<const v8bf*>(p);
  v8bf hi = *reinterpret_cast<const v8bf*>(p + 8);
  v16bf r;
#pragma unroll
  for (int i = 0; i < 8; ++i) { r[i] = lo[i]; r[i + 8] = hi[i]; }
  return r;
}

// ---------------- Tensor Data Mover: 2-D tile (tile_x contiguous elems per line,
// tile_y lines, line stride = stride_elems) global -> LDS, bf16 elements ----------------
__device__ __forceinline__ void tdm_load_2d(unsigned lds_off, const void* gptr,
                                            unsigned tile_x, unsigned tile_y,
                                            unsigned stride_elems) {
  unsigned long long ga = (unsigned long long)(uintptr_t)gptr;
  v4u g0;
  g0[0] = 1u;                                  // count=1, user mode
  g0[1] = lds_off;                             // lds_addr (bytes)
  g0[2] = (unsigned)ga;                        // global_addr[31:0]
  g0[3] = (unsigned)(ga >> 32) | (2u << 30);   // global_addr[56:32] | type=2
  v8u g1;
  g1[0] = 0x00010000u;                         // workgroup_mask=0, data_size=1 (2B)
  g1[1] = 0xFFFF0000u;                         // tensor_dim0[15:0]=0xFFFF (huge, no clip)
  g1[2] = 0x7FFFu | 0xFFFF0000u;               // tensor_dim0[31:16] | tensor_dim1[15:0]
  g1[3] = 0x7FFFu | (tile_x << 16);            // tensor_dim1[31:16] | tile_dim0
  g1[4] = tile_y & 0xFFFFu;                    // tile_dim1 | tile_dim2=0
  g1[5] = stride_elems;                        // tensor_dim0_stride[31:0]
  g1[6] = 0u;                                  // stride0[47:32] | stride1[15:0]
  g1[7] = 0u;
  v4u g2 = {0u, 0u, 0u, 0u};                   // tensor_dim2=0, tile_dim3=0
  v4u g3 = {0u, 0u, 0u, 0u};
  asm volatile("tensor_load_to_lds %0, %1, %2, %3"
               :: "s"(g0), "s"(g1), "s"(g2), "s"(g3)
               : "memory");
}

// ---------------- kernel 1: f32 -> bf16 cast ----------------
__global__ __launch_bounds__(BDIM) void cast_kernel(const float* __restrict__ in,
                                                    bf16_t* __restrict__ out, int n4) {
  int i = blockIdx.x * blockDim.x + threadIdx.x;
  if (i >= n4) return;
  const float4 v = reinterpret_cast<const float4*>(in)[i];
  bf16_t* o = out + (size_t)i * 4;
  o[0] = f2bf(v.x); o[1] = f2bf(v.y); o[2] = f2bf(v.z); o[3] = f2bf(v.w);
}

// ---------------- kernel 2: fused QKV projection (y = x @ W^T) ----------------
// grid: (Dn/128, Mtot/128, 3); block 256 = 8 waves, wave tile 32x64.
// W tile (128 n-rows x 32 k) staged via TDM into LDS [n][32].
__global__ __launch_bounds__(BDIM) void qkv_gemm_kernel(const bf16_t* __restrict__ xb,
                                                        const bf16_t* __restrict__ wb_all,
                                                        bf16_t* __restrict__ qkv_all) {
  const int z = blockIdx.z;
  const bf16_t* W  = wb_all + (size_t)z * Dn * Dn;
  bf16_t* out      = qkv_all + (size_t)z * Mtot * Dn;
  const float scale = (z == 0) ? 0.03125f : 1.0f;   // fold 1/sqrt(1024) into Q

  const int wave = threadIdx.x >> 5;
  const int wm = wave >> 1, wn = wave & 1;
  const int row0 = blockIdx.y * 128 + wm * 32;
  const int nblk = blockIdx.x * 128;
  const int col0 = nblk + wn * 64;
  const int lane = threadIdx.x & 31;

  __shared__ bf16_t ldsW[128 * 32];   // [n_local][k], pitch 32 (64B rows, 16B aligned)
  const unsigned lds_off = (unsigned)(uintptr_t)(&ldsW[0]);

  v8f acc[2][4] = {};
  for (int kb = 0; kb < Dn; kb += 32) {
    if (wave == 0) {
      tdm_load_2d(lds_off, W + (size_t)nblk * Dn + kb, 32u, 128u, (unsigned)Dn);
      __builtin_amdgcn_s_wait_tensorcnt(0);
    }
    __syncthreads();

    __builtin_prefetch(xb + (size_t)row0 * Dn + kb + 64, 0, 1);
    v16bf a[2], b[4];
#pragma unroll
    for (int i = 0; i < 2; ++i)
      a[i] = load_a_frag(xb + (size_t)(row0 + i * 16) * Dn + kb, Dn);
#pragma unroll
    for (int j = 0; j < 4; ++j) {
      const bf16_t* bp = &ldsW[(size_t)(wn * 64 + j * 16 + (lane & 15)) * 32 + ((lane >> 4) << 4)];
      v8bf lo = *reinterpret_cast<const v8bf*>(bp);
      v8bf hi = *reinterpret_cast<const v8bf*>(bp + 8);
      v16bf r;
#pragma unroll
      for (int e = 0; e < 8; ++e) { r[e] = lo[e]; r[e + 8] = hi[e]; }
      b[j] = r;
    }
#pragma unroll
    for (int i = 0; i < 2; ++i)
#pragma unroll
      for (int j = 0; j < 4; ++j)
        acc[i][j] = __builtin_amdgcn_wmma_f32_16x16x32_bf16(
            false, a[i], false, b[j], (short)0, acc[i][j], false, false);
    __syncthreads();
  }
  const int cc = lane & 15, rb = (lane >> 4) << 3;
#pragma unroll
  for (int i = 0; i < 2; ++i)
#pragma unroll
    for (int j = 0; j < 4; ++j)
#pragma unroll
      for (int v = 0; v < 8; ++v)
        out[(size_t)(row0 + i * 16 + rb + v) * Dn + (col0 + j * 16 + cc)] =
            f2bf(acc[i][j][v] * scale);
}

// ---------------- kernel 3: scores = Q @ K^T (causal tile skip) ----------------
// grid: (Sn/128, Sn/128, Bn). K tile staged via TDM into LDS [n][32].
__global__ __launch_bounds__(BDIM) void scores_gemm_kernel(const bf16_t* __restrict__ qbuf,
                                                           const bf16_t* __restrict__ kbuf,
                                                           float* __restrict__ scores) {
  if (blockIdx.x > blockIdx.y) return;   // tile entirely above the diagonal
  const int b = blockIdx.z;
  const int wave = threadIdx.x >> 5;
  const int wm = wave >> 1, wn = wave & 1;
  const int q0 = blockIdx.y * 128 + wm * 32;
  const int kblk = blockIdx.x * 128;
  const int k0 = kblk + wn * 64;
  const bf16_t* Q = qbuf + (size_t)b * Sn * Dn;
  const bf16_t* K = kbuf + (size_t)b * Sn * Dn;
  float* Sout = scores + (size_t)b * Sn * Sn;
  const int lane = threadIdx.x & 31;

  __shared__ bf16_t ldsK[128 * 32];
  const unsigned lds_off = (unsigned)(uintptr_t)(&ldsK[0]);

  v8f acc[2][4] = {};
  for (int kb = 0; kb < Dn; kb += 32) {
    if (wave == 0) {
      tdm_load_2d(lds_off, K + (size_t)kblk * Dn + kb, 32u, 128u, (unsigned)Dn);
      __builtin_amdgcn_s_wait_tensorcnt(0);
    }
    __syncthreads();

    __builtin_prefetch(Q + (size_t)q0 * Dn + kb + 64, 0, 1);
    v16bf a[2], bb[4];
#pragma unroll
    for (int i = 0; i < 2; ++i)
      a[i] = load_a_frag(Q + (size_t)(q0 + i * 16) * Dn + kb, Dn);
#pragma unroll
    for (int j = 0; j < 4; ++j) {
      const bf16_t* bp = &ldsK[(size_t)(wn * 64 + j * 16 + (lane & 15)) * 32 + ((lane >> 4) << 4)];
      v8bf lo = *reinterpret_cast<const v8bf*>(bp);
      v8bf hi = *reinterpret_cast<const v8bf*>(bp + 8);
      v16bf r;
#pragma unroll
      for (int e = 0; e < 8; ++e) { r[e] = lo[e]; r[e + 8] = hi[e]; }
      bb[j] = r;
    }
#pragma unroll
    for (int i = 0; i < 2; ++i)
#pragma unroll
      for (int j = 0; j < 4; ++j)
        acc[i][j] = __builtin_amdgcn_wmma_f32_16x16x32_bf16(
            false, a[i], false, bb[j], (short)0, acc[i][j], false, false);
    __syncthreads();
  }
  const int cc = lane & 15, rb = (lane >> 4) << 3;
#pragma unroll
  for (int i = 0; i < 2; ++i)
#pragma unroll
    for (int j = 0; j < 4; ++j)
#pragma unroll
      for (int v = 0; v < 8; ++v)
        Sout[(size_t)(q0 + i * 16 + rb + v) * Sn + (k0 + j * 16 + cc)] = acc[i][j][v];
}

// ---------------- kernel 4: causal softmax, in-place f32 row -> bf16 P row ----------------
__global__ __launch_bounds__(BDIM) void softmax_kernel(float* __restrict__ scores) {
  const int row = blockIdx.x;          // 0..Mtot-1
  const int q = row & (Sn - 1);
  const int L = q + 1;                 // causal length
  float* srow = scores + (size_t)row * Sn;
  __shared__ float buf[Sn];
  __shared__ float red[BDIM];
  const int t = threadIdx.x;

  float m = -3.4e38f;
  for (int i = t; i < Sn; i += BDIM) {
    float v = (i < L) ? srow[i] : -3.4e38f;
    buf[i] = v;
    m = fmaxf(m, v);
  }
  red[t] = m; __syncthreads();
  for (int s = BDIM / 2; s > 0; s >>= 1) {
    if (t < s) red[t] = fmaxf(red[t], red[t + s]);
    __syncthreads();
  }
  m = red[0]; __syncthreads();

  float sum = 0.f;
  for (int i = t; i < Sn; i += BDIM) {
    float e = (i < L) ? __expf(buf[i] - m) : 0.f;
    buf[i] = e; sum += e;
  }
  red[t] = sum; __syncthreads();
  for (int s = BDIM / 2; s > 0; s >>= 1) {
    if (t < s) red[t] += red[t + s];
    __syncthreads();
  }
  const float inv = 1.0f / red[0];

  bf16_t* prow = reinterpret_cast<bf16_t*>(srow);  // in-place within this row's 16KB
  for (int i = t; i < Sn; i += BDIM) prow[i] = f2bf(buf[i] * inv);
}

// ---------------- kernel 5: out = P @ V  (V transposed through LDS manually) ----------------
// grid: (Dn/128, Mtot/128)
__global__ __launch_bounds__(BDIM) void out_gemm_kernel(const float* __restrict__ scores,
                                                        const bf16_t* __restrict__ vbuf,
                                                        float* __restrict__ out) {
  const int wave = threadIdx.x >> 5;
  const int wm = wave >> 1, wn = wave & 1;
  const int row0g = blockIdx.y * 128;
  const int b = row0g / Sn;
  const int row0 = row0g + wm * 32;
  const int col0 = blockIdx.x * 128 + wn * 64;
  const bf16_t* P = reinterpret_cast<const bf16_t*>(scores);  // row pitch 2*Sn bf16
  const int ldp = 2 * Sn;
  const bf16_t* V = vbuf + (size_t)b * Sn * Dn;

  __shared__ bf16_t ldsB[128 * 40];   // [n_local 0..127][k 0..31], pitch 40 (pad)
  const int t = threadIdx.x;
  const int lane = t & 31;

  v8f acc[2][4] = {};
  for (int kb = 0; kb < Sn; kb += 32) {
    // stage V[kb..kb+31][nb..nb+127] transposed into LDS (coalesced global reads)
    {
      const int kk = t >> 4;           // 0..15
      const int cols = (t & 15) * 8;
      const int nb = blockIdx.x * 128;
#pragma unroll
      for (int h = 0; h < 2; ++h) {
        const int krow = kk + h * 16;
        v8bf vv = *reinterpret_cast<const v8bf*>(V + (size_t)(kb + krow) * Dn + nb + cols);
#pragma unroll
        for (int e = 0; e < 8; ++e) ldsB[(cols + e) * 40 + krow] = vv[e];
      }
    }
    __syncthreads();

    v16bf a[2], bfrag[4];
#pragma unroll
    for (int i = 0; i < 2; ++i)
      a[i] = load_a_frag(P + (size_t)(row0 + i * 16) * ldp + kb, ldp);
#pragma unroll
    for (int j = 0; j < 4; ++j) {
      const bf16_t* bp = &ldsB[(size_t)(wn * 64 + j * 16 + (lane & 15)) * 40 + ((lane >> 4) << 4)];
      v8bf lo = *reinterpret_cast<const v8bf*>(bp);
      v8bf hi = *reinterpret_cast<const v8bf*>(bp + 8);
      v16bf r;
#pragma unroll
      for (int e = 0; e < 8; ++e) { r[e] = lo[e]; r[e + 8] = hi[e]; }
      bfrag[j] = r;
    }
#pragma unroll
    for (int i = 0; i < 2; ++i)
#pragma unroll
      for (int j = 0; j < 4; ++j)
        acc[i][j] = __builtin_amdgcn_wmma_f32_16x16x32_bf16(
            false, a[i], false, bfrag[j], (short)0, acc[i][j], false, false);
    __syncthreads();
  }
  const int cc = lane & 15, rb = (lane >> 4) << 3;
#pragma unroll
  for (int i = 0; i < 2; ++i)
#pragma unroll
    for (int j = 0; j < 4; ++j)
#pragma unroll
      for (int v = 0; v < 8; ++v)
        out[(size_t)(row0 + i * 16 + rb + v) * Dn + (col0 + j * 16 + cc)] = acc[i][j][v];
}

// ---------------- host launcher ----------------
extern "C" void kernel_launch(void* const* d_in, const int* in_sizes, int n_in,
                              void* d_out, int out_size, void* d_ws, size_t ws_size,
                              hipStream_t stream) {
  const float* x  = (const float*)d_in[0];
  const float* Wq = (const float*)d_in[1];
  const float* Wk = (const float*)d_in[2];
  const float* Wv = (const float*)d_in[3];

  char* ws = (char*)d_ws;
  size_t off = 0;
  bf16_t* xb  = (bf16_t*)(ws + off); off += (size_t)Mtot * Dn * 2;        // 16.8 MB
  bf16_t* wb  = (bf16_t*)(ws + off); off += 3ull * Dn * Dn * 2;           //  6.3 MB
  bf16_t* qkv = (bf16_t*)(ws + off); off += 3ull * Mtot * Dn * 2;         // 50.3 MB
  float*  sc  = (float*)(ws + off);  off += (size_t)Bn * Sn * Sn * 4;     // 134 MB (L2-resident)

  // 1) casts to bf16
  {
    int n4 = Mtot * Dn / 4;
    cast_kernel<<<(n4 + BDIM - 1) / BDIM, BDIM, 0, stream>>>(x, xb, n4);
    int w4 = Dn * Dn / 4;
    cast_kernel<<<(w4 + BDIM - 1) / BDIM, BDIM, 0, stream>>>(Wq, wb + 0ull * Dn * Dn, w4);
    cast_kernel<<<(w4 + BDIM - 1) / BDIM, BDIM, 0, stream>>>(Wk, wb + 1ull * Dn * Dn, w4);
    cast_kernel<<<(w4 + BDIM - 1) / BDIM, BDIM, 0, stream>>>(Wv, wb + 2ull * Dn * Dn, w4);
  }

  // 2) fused QKV projection (TDM-staged weights)
  qkv_gemm_kernel<<<dim3(Dn / 128, Mtot / 128, 3), BDIM, 0, stream>>>(xb, wb, qkv);

  bf16_t* qb = qkv;
  bf16_t* kb = qkv + 1ull * Mtot * Dn;
  bf16_t* vb = qkv + 2ull * Mtot * Dn;

  // 3) scores (causal-skipped, TDM-staged K tiles)
  scores_gemm_kernel<<<dim3(Sn / 128, Sn / 128, Bn), BDIM, 0, stream>>>(qb, kb, sc);

  // 4) softmax -> P (bf16, in place)
  softmax_kernel<<<Mtot, BDIM, 0, stream>>>(sc);

  // 5) out = P @ V
  out_gemm_kernel<<<dim3(Dn / 128, Mtot / 128), BDIM, 0, stream>>>(sc, vb, (float*)d_out);
}
